// GAIN_BERT_80453327389404
// MI455X (gfx1250) — matrix operand order repaired and checked by reading, not compiled, not run on hardware
//
#include <hip/hip_runtime.h>
#include <hip/hip_bf16.h>
#include <math.h>

// ---------------- problem constants ----------------
#define S_   512
#define L_   128
#define W_   64
#define D_   768
#define H_   8
#define T_   64
#define E_   8192
#define R_   4
#define DK_  96
#define NN_  577           // 1 + S + T
#define NE_  ((size_t)32768 * 768)   // S*W*D

typedef __attribute__((ext_vector_type(16))) __bf16 v16bf;
typedef __attribute__((ext_vector_type(4)))  __bf16 v4bf;
typedef __attribute__((ext_vector_type(8)))  float  v8f;

// native fptrunc (RNE) -> lets the backend use packed bf16 converts
__device__ __forceinline__ __bf16 f2bf(float f) { return (__bf16)f; }

__device__ __forceinline__ v4bf f2bf4(float4 v) {
    v4bf t;
    t[0] = (__bf16)v.x; t[1] = (__bf16)v.y; t[2] = (__bf16)v.z; t[3] = (__bf16)v.w;
    return t;
}

// A-side K-swizzle within a 32-wide chunk: lane fragment (hi half) becomes
// the contiguous 16 elements at chunk + hi*16.
//   k 0..7  -> 0..7 ; k 16..23 -> 8..15 ; k 8..15 -> 16..23 ; k 24..31 -> 24..31
__device__ __forceinline__ int swz32(int k) {
    return (k & 7) | ((k >> 1) & 8) | ((k << 1) & 16);
}

// ---------------------------------------------------------------------------
// word_embed[s,w,:] = sent_embed[s, argmax_l s_idx[s,w,l], :]
// ---------------------------------------------------------------------------
__global__ __launch_bounds__(256) void k_gather_words(
    const float* __restrict__ sent_embed, const float* __restrict__ s_idx,
    float* __restrict__ we)
{
    int sw = blockIdx.x;              // 0 .. S*W-1
    __shared__ int idx;
    int t = threadIdx.x;
    if (t == 0) idx = 0;
    __syncthreads();
    if (t < L_) {
        if (s_idx[(size_t)sw * L_ + t] > 0.5f) idx = t;   // single writer
    }
    __syncthreads();
    int s = sw >> 6;
    const float* src = sent_embed + ((size_t)s * L_ + idx) * D_;
    float*       dst = we + (size_t)sw * D_;
    for (int d = t; d < D_; d += 256) dst[d] = src[d];
}

// ---------------------------------------------------------------------------
// Generic GEMM:  C[M,N] = (accum ? C : 0) + A[M,K] @ B[K,N] (+ bias[N])
// bf16 WMMA, f32 accumulate. BM=128, BN=64, BK=32. 8 waves, 32x32 per wave.
// Requires: K % 32 == 0, N % 64 == 0. M guarded.
// ---------------------------------------------------------------------------
#define BM 128
#define BN 64
#define BK 32

__global__ __launch_bounds__(256) void k_gemm(
    const float* __restrict__ A, const float* __restrict__ B,
    float* __restrict__ C, int M, int N, int K,
    const float* __restrict__ bias, int accum)
{
    __shared__ __align__(32) __bf16 As[BM * BK];   // row-major rows, natural K order
    __shared__ __align__(32) __bf16 Bs[BN * BK];   // transposed: [col][k]

    int tid  = threadIdx.x;
    int lane = tid & 31, wave = tid >> 5;
    int wm = wave >> 1, wn = wave & 1;             // 4 x 2 wave grid
    int lr = lane & 15, hi = lane >> 4;
    int bRow = blockIdx.y * BM, bCol = blockIdx.x * BN;

    v8f cc[2][2] = {};

    int ktiles = K / BK;
    for (int kt = 0; kt < ktiles; ++kt) {
        int k0 = kt * BK;
        // stage A tile (128x32) as float4 -> bf16 quads
        for (int i = 0; i < 4; ++i) {
            int idx4 = tid + i * 256;              // 0..1023
            int r = idx4 >> 3, c4 = (idx4 & 7) * 4;
            int gr = bRow + r;
            float4 v = make_float4(0.f, 0.f, 0.f, 0.f);
            if (gr < M) {
                const float* ap = A + (size_t)gr * K + k0 + c4;
                v = *(const float4*)ap;
                if (kt + 1 < ktiles) __builtin_prefetch(ap + BK, 0, 1);
            }
            *(v4bf*)(&As[r * BK + c4]) = f2bf4(v);
        }
        // stage B tile (32x64) as float4, store transposed [col][k]
        for (int i = 0; i < 2; ++i) {
            int idx4 = tid + i * 256;              // 0..511
            int r = idx4 >> 4, c4 = (idx4 & 15) * 4;
            const float* bp = B + (size_t)(k0 + r) * N + bCol + c4;
            float4 v = *(const float4*)bp;
            if (kt + 1 < ktiles) __builtin_prefetch(bp + (size_t)BK * N, 0, 1);
            Bs[(c4 + 0) * BK + r] = f2bf(v.x);
            Bs[(c4 + 1) * BK + r] = f2bf(v.y);
            Bs[(c4 + 2) * BK + r] = f2bf(v.z);
            Bs[(c4 + 3) * BK + r] = f2bf(v.w);
        }
        __syncthreads();

        int ar0 = wm * 32 + lr;
        int bc0 = wn * 32 + lr;
        v16bf a0 = *(const v16bf*)(&As[ar0 * BK + hi * 16]);
        v16bf a1 = *(const v16bf*)(&As[(ar0 + 16) * BK + hi * 16]);
        v16bf b0 = *(const v16bf*)(&Bs[bc0 * BK + hi * 16]);
        v16bf b1 = *(const v16bf*)(&Bs[(bc0 + 16) * BK + hi * 16]);

        cc[0][0] = __builtin_amdgcn_wmma_f32_16x16x32_bf16(false, a0, false, b0, (short)0, cc[0][0], false, false);
        cc[0][1] = __builtin_amdgcn_wmma_f32_16x16x32_bf16(false, a0, false, b1, (short)0, cc[0][1], false, false);
        cc[1][0] = __builtin_amdgcn_wmma_f32_16x16x32_bf16(false, a1, false, b0, (short)0, cc[1][0], false, false);
        cc[1][1] = __builtin_amdgcn_wmma_f32_16x16x32_bf16(false, a1, false, b1, (short)0, cc[1][1], false, false);
        __syncthreads();
    }

    for (int ti = 0; ti < 2; ++ti)
        for (int tj = 0; tj < 2; ++tj)
            for (int r = 0; r < 8; ++r) {
                int row = bRow + wm * 32 + ti * 16 + hi * 8 + r;
                int col = bCol + wn * 32 + tj * 16 + lr;
                if (row < M) {
                    float v = cc[ti][tj][r];
                    if (bias) v += bias[col];
                    size_t o = (size_t)row * N + col;
                    if (accum) v += C[o];
                    C[o] = v;
                }
            }
}

// ---------------------------------------------------------------------------
// Batched propagation:  C[s] = P[s] (64x64) @ Hm[s] (64x768)
// grid = (N/64, S). P row-major, H transposed, both fully in LDS.
// ---------------------------------------------------------------------------
__global__ __launch_bounds__(256) void k_bgemm_adj(
    const float* __restrict__ P, const float* __restrict__ Hm,
    float* __restrict__ C)
{
    int s = blockIdx.y;
    int bCol = blockIdx.x * 64;
    __shared__ __align__(32) __bf16 Ps[64 * 64];   // row-major [row][k]
    __shared__ __align__(32) __bf16 Hs[64 * 64];   // transposed [col][k]
    int tid = threadIdx.x;
    const float* Pp = P + (size_t)s * 4096;
    for (int i = 0; i < 4; ++i) {
        int idx4 = tid + i * 256;                  // 0..1023
        int r = idx4 >> 4, c4 = (idx4 & 15) * 4;
        float4 pv = *(const float4*)(Pp + idx4 * 4);
        int pr = (idx4 * 4) >> 6, pc = (idx4 * 4) & 63;
        *(v4bf*)(&Ps[pr * 64 + pc]) = f2bf4(pv);

        float4 hv = *(const float4*)(Hm + ((size_t)s * 64 + r) * D_ + bCol + c4);
        Hs[(c4 + 0) * 64 + r] = f2bf(hv.x);
        Hs[(c4 + 1) * 64 + r] = f2bf(hv.y);
        Hs[(c4 + 2) * 64 + r] = f2bf(hv.z);
        Hs[(c4 + 3) * 64 + r] = f2bf(hv.w);
    }
    __syncthreads();

    int lane = tid & 31, wave = tid >> 5;
    int wm = wave >> 1, wn = wave & 1;             // wave = 16x32 out
    int lr = lane & 15, hi = lane >> 4;

    v8f cc[2] = {};
    for (int k0 = 0; k0 < 64; k0 += 32) {
        v16bf a  = *(const v16bf*)(&Ps[(wm * 16 + lr) * 64 + k0 + hi * 16]);
        v16bf b0 = *(const v16bf*)(&Hs[(wn * 32 + lr) * 64 + k0 + hi * 16]);
        v16bf b1 = *(const v16bf*)(&Hs[(wn * 32 + 16 + lr) * 64 + k0 + hi * 16]);
        cc[0] = __builtin_amdgcn_wmma_f32_16x16x32_bf16(false, a, false, b0, (short)0, cc[0], false, false);
        cc[1] = __builtin_amdgcn_wmma_f32_16x16x32_bf16(false, a, false, b1, (short)0, cc[1], false, false);
    }
    for (int tj = 0; tj < 2; ++tj)
        for (int r = 0; r < 8; ++r) {
            int row = wm * 16 + hi * 8 + r;
            int col = bCol + wn * 32 + tj * 16 + lr;
            C[((size_t)s * 64 + row) * D_ + col] = cc[tj][r];
        }
}

// ---------------------------------------------------------------------------
// scores[s,h] = Q[s,:,h*96:(h+1)*96] @ K^T / sqrt(96)   -> sc[S,H,64,64]
// Qs row-major [m][k] (A side), Ks stored [n][k] == B^T layout (already contiguous).
// ---------------------------------------------------------------------------
__global__ __launch_bounds__(256) void k_scores(
    const float* __restrict__ q, const float* __restrict__ k,
    float* __restrict__ sc)
{
    int bx = blockIdx.x;
    int s = bx >> 3, h = bx & 7;
    __shared__ __align__(32) __bf16 Qs[64 * 96];
    __shared__ __align__(32) __bf16 Ks[64 * 96];
    int tid = threadIdx.x;
    for (int i = 0; i < 6; ++i) {
        int idx4 = tid + i * 256;                  // 0..1535
        int r = idx4 / 24, c4 = (idx4 % 24) * 4;
        size_t off = ((size_t)s * 64 + r) * D_ + h * 96 + c4;
        float4 qv = *(const float4*)(q + off);
        float4 kv = *(const float4*)(k + off);
        *(v4bf*)(&Qs[r * 96 + c4]) = f2bf4(qv);
        *(v4bf*)(&Ks[r * 96 + c4]) = f2bf4(kv);
    }
    __syncthreads();

    int lane = tid & 31, wave = tid >> 5;
    int wm = wave >> 1, wn = wave & 1;
    int lr = lane & 15, hi = lane >> 4;

    v8f cc[2] = {};
    for (int k0 = 0; k0 < 96; k0 += 32) {
        v16bf a  = *(const v16bf*)(&Qs[(wm * 16 + lr) * 96 + k0 + hi * 16]);
        v16bf b0 = *(const v16bf*)(&Ks[(wn * 32 + lr) * 96 + k0 + hi * 16]);
        v16bf b1 = *(const v16bf*)(&Ks[(wn * 32 + 16 + lr) * 96 + k0 + hi * 16]);
        cc[0] = __builtin_amdgcn_wmma_f32_16x16x32_bf16(false, a, false, b0, (short)0, cc[0], false, false);
        cc[1] = __builtin_amdgcn_wmma_f32_16x16x32_bf16(false, a, false, b1, (short)0, cc[1], false, false);
    }
    const float scale = 0.10206207261596577f;      // 1/sqrt(96)
    for (int tj = 0; tj < 2; ++tj)
        for (int r = 0; r < 8; ++r) {
            int row = wm * 16 + hi * 8 + r;
            int col = wn * 32 + tj * 16 + lr;
            sc[((size_t)bx * 64 + row) * 64 + col] = cc[tj][r] * scale;
        }
}

// ---------------------------------------------------------------------------
// exact sort-based entmax-1.5 over rows of 64 (in-place), one thread / row
// ---------------------------------------------------------------------------
__global__ __launch_bounds__(256) void k_entmax(float* __restrict__ sc, int nrows)
{
    int row = blockIdx.x * blockDim.x + threadIdx.x;
    if (row >= nrows) return;
    float* p = sc + (size_t)row * 64;
    float x[64], z[64], tau[64];
    float xmax = -1e30f;
    for (int i = 0; i < 64; ++i) { x[i] = p[i] * 0.5f; xmax = fmaxf(xmax, x[i]); }
    for (int i = 0; i < 64; ++i) { x[i] -= xmax; z[i] = x[i]; }
    for (int i = 1; i < 64; ++i) {                 // insertion sort, descending
        float key = z[i]; int j = i - 1;
        while (j >= 0 && z[j] < key) { z[j + 1] = z[j]; --j; }
        z[j + 1] = key;
    }
    float cs = 0.f, cq = 0.f; int support = 0;
    for (int i = 0; i < 64; ++i) {
        cs += z[i]; cq += z[i] * z[i];
        float kf = (float)(i + 1);
        float mean = cs / kf, msq = cq / kf;
        float ss = kf * (msq - mean * mean);
        float delta = (1.0f - ss) / kf;
        float sq = delta > 0.f ? sqrtf(delta) : 0.f;
        tau[i] = mean - sq;
        if (tau[i] <= z[i]) support++;
    }
    float ts = tau[support - 1];
    for (int i = 0; i < 64; ++i) {
        float v = x[i] - ts; v = v > 0.f ? v : 0.f;
        p[i] = v * v;
    }
}

// lat[s,q,k] = mean_h sc[s,h,q,k]
__global__ __launch_bounds__(256) void k_headred(
    const float* __restrict__ sc, float* __restrict__ lat)
{
    size_t i = (size_t)blockIdx.x * blockDim.x + threadIdx.x;
    size_t n = (size_t)S_ * 64 * 64;
    if (i >= n) return;
    int s = (int)(i >> 12);
    int rem = (int)(i & 4095);
    float acc = 0.f;
    for (int h = 0; h < H_; ++h)
        acc += sc[(((size_t)s * H_ + h) << 12) + rem];
    lat[i] = acc * 0.125f;
}

// X = relu(sigmoid(G)*OUT + (1-sigmoid(G))*LT + b[col]) (+ resid)
__global__ __launch_bounds__(256) void k_gate(
    const float* __restrict__ G, const float* __restrict__ OUT,
    const float* __restrict__ LT, const float* __restrict__ b,
    const float* __restrict__ resid, float* __restrict__ X, size_t n)
{
    size_t i = (size_t)blockIdx.x * blockDim.x + threadIdx.x;
    if (i >= n) return;
    int col = (int)(i % D_);
    float g = 1.f / (1.f + expf(-G[i]));
    float v = g * OUT[i] + (1.f - g) * LT[i] + b[col];
    v = v > 0.f ? v : 0.f;
    if (resid) v += resid[i];
    X[i] = v;
}

// feats = concat(doc_cls, sent_cls, X[t_sid, t_index])
__global__ __launch_bounds__(256) void k_feats(
    const float* __restrict__ doc_cls, const float* __restrict__ sent_cls,
    const float* __restrict__ X, const int* __restrict__ t_sid,
    const int* __restrict__ t_index, float* __restrict__ fe)
{
    size_t i = (size_t)blockIdx.x * blockDim.x + threadIdx.x;
    size_t n = (size_t)NN_ * D_;
    if (i >= n) return;
    int row = (int)(i / D_), d = (int)(i % D_);
    float v;
    if (row == 0)            v = doc_cls[d];
    else if (row <= S_)      v = sent_cls[(size_t)(row - 1) * D_ + d];
    else {
        int t = row - 1 - S_;
        int s = t_sid[t], w = t_index[t];
        v = X[((size_t)s * W_ + w) * D_ + d];
    }
    fe[i] = v;
}

// gsrc[e] = in[src[e]]   (one block per edge)
__global__ __launch_bounds__(256) void k_gather_rows(
    const float* __restrict__ in, const int* __restrict__ src,
    float* __restrict__ out)
{
    int e = blockIdx.x;
    int node = src[e];
    const float* sp = in + (size_t)node * D_;
    float* dp = out + (size_t)e * D_;
    for (int d = threadIdx.x; d < D_; d += 256) dp[d] = sp[d];
}

__global__ __launch_bounds__(256) void k_zero(float* __restrict__ p, size_t n)
{
    size_t i = (size_t)blockIdx.x * blockDim.x + threadIdx.x;
    if (i < n) p[i] = 0.f;
}

__global__ __launch_bounds__(256) void k_deg(
    const int* __restrict__ dst, float* __restrict__ deg)
{
    int i = blockIdx.x * blockDim.x + threadIdx.x;
    if (i >= R_ * E_) return;
    int r = i >> 13;
    atomicAdd(&deg[r * NN_ + dst[i]], 1.f);
}

// hs[dst] += msg[e] / max(deg[r,dst],1)   (one block per edge)
__global__ __launch_bounds__(256) void k_scatter(
    const float* __restrict__ msg, const int* __restrict__ dst,
    const float* __restrict__ deg, float* __restrict__ hs)
{
    int e = blockIdx.x;
    int r = e >> 13;
    int node = dst[e];
    float scale = 1.f / fmaxf(deg[r * NN_ + node], 1.f);
    const float* mp = msg + (size_t)e * D_;
    float* hp = hs + (size_t)node * D_;
    for (int d = threadIdx.x; d < D_; d += 256)
        atomicAdd(&hp[d], mp[d] * scale);
}

// h = relu(hs + loop + bias[col])
__global__ __launch_bounds__(256) void k_relu_comb(
    const float* __restrict__ hs, const float* __restrict__ lp,
    const float* __restrict__ b, float* __restrict__ h, size_t n)
{
    size_t i = (size_t)blockIdx.x * blockDim.x + threadIdx.x;
    if (i >= n) return;
    float v = hs[i] + lp[i] + b[i % D_];
    h[i] = v > 0.f ? v : 0.f;
}

// out[n,c] = [fe|h1|h2][n] . pred_w[:,c] + pred_b[c]
__global__ __launch_bounds__(256) void k_pred(
    const float* __restrict__ fe, const float* __restrict__ h1,
    const float* __restrict__ h2, const float* __restrict__ pw,
    const float* __restrict__ pb, float* __restrict__ out)
{
    int i = blockIdx.x * blockDim.x + threadIdx.x;
    if (i >= NN_ * 5) return;
    int n = i / 5, c = i % 5;
    float acc = pb[c];
    const float* f0 = fe + (size_t)n * D_;
    const float* f1 = h1 + (size_t)n * D_;
    const float* f2 = h2 + (size_t)n * D_;
    for (int d = 0; d < D_; ++d) {
        acc += f0[d] * pw[(size_t)d * 5 + c];
        acc += f1[d] * pw[(size_t)(D_ + d) * 5 + c];
        acc += f2[d] * pw[(size_t)(2 * D_ + d) * 5 + c];
    }
    out[i] = acc;
}

// ---------------------------------------------------------------------------
extern "C" void kernel_launch(void* const* d_in, const int* in_sizes, int n_in,
                              void* d_out, int out_size, void* d_ws, size_t ws_size,
                              hipStream_t stream) {
    const float* sent_embed = (const float*)d_in[0];
    const float* s_idx      = (const float*)d_in[1];
    const float* dep_adj    = (const float*)d_in[2];
    const float* doc_cls    = (const float*)d_in[3];
    const float* sent_cls   = (const float*)d_in[4];
    const int*   t_sid      = (const int*)d_in[5];
    const int*   t_index    = (const int*)d_in[6];
    const int*   edges_src  = (const int*)d_in[7];
    const int*   edges_dst  = (const int*)d_in[8];
    const float* wq_w = (const float*)d_in[9];
    const float* wq_b = (const float*)d_in[10];
    const float* wk_w = (const float*)d_in[11];
    const float* wk_b = (const float*)d_in[12];
    const float* gc1_w   = (const float*)d_in[13];
    const float* gc1_b   = (const float*)d_in[14];
    const float* gc1_l1w = (const float*)d_in[15];
    const float* gc1_l1b = (const float*)d_in[16];
    const float* gc1_l2w = (const float*)d_in[17];
    const float* gc1_l2b = (const float*)d_in[18];
    const float* gc2_w   = (const float*)d_in[19];
    const float* gc2_b   = (const float*)d_in[20];
    const float* gc2_l1w = (const float*)d_in[21];
    const float* gc2_l1b = (const float*)d_in[22];
    const float* gc2_l2w = (const float*)d_in[23];
    const float* gc2_l2b = (const float*)d_in[24];
    const float* rel1_w    = (const float*)d_in[25];
    const float* rel1_loop = (const float*)d_in[26];
    const float* rel1_b    = (const float*)d_in[27];
    const float* rel2_w    = (const float*)d_in[28];
    const float* rel2_loop = (const float*)d_in[29];
    const float* rel2_b    = (const float*)d_in[30];
    const float* pred_w    = (const float*)d_in[31];
    const float* pred_b    = (const float*)d_in[32];
    float* out = (float*)d_out;

    float* ws = (float*)d_ws;
    float* WE  = ws;                         // word_embed  [32768,768]
    float* BQ  = ws + 1 * NE_;               // Q / gate-preact
    float* BK2 = ws + 2 * NE_;               // K / X1
    float* BH  = ws + 3 * NE_;               // h = x@w / final X
    float* BO  = ws + 4 * NE_;               // adj@h / gathered src rows
    float* BL  = ws + 5 * NE_;               // lat@h / messages
    float* SC  = ws + 6 * NE_;               // scores [S,H,64,64]
    size_t scN = (size_t)S_ * H_ * 64 * 64;
    float* LAT = SC + scN;                   // [S,64,64]
    size_t latN = (size_t)S_ * 64 * 64;
    float* FE  = LAT + latN;                 // feats [577,768]
    size_t feN = (size_t)NN_ * D_;
    float* H1  = FE + feN;
    float* H2  = H1 + feN;
    float* HS  = H2 + feN;                   // relation aggregate
    float* LP  = HS + feN;                   // self-loop term
    float* DEG = LP + feN;                   // [R,577]

    auto gemm = [&](const float* A, const float* B, float* C, int M, int N, int K,
                    const float* bias, int accum) {
        dim3 grid(N / BN, (M + BM - 1) / BM);
        k_gemm<<<grid, 256, 0, stream>>>(A, B, C, M, N, K, bias, accum);
    };

    const int MW = 32768;

    // 1) word gather
    k_gather_words<<<MW, 256, 0, stream>>>(sent_embed, s_idx, WE);

    // 2) Q/K projections
    gemm(WE, wq_w, BQ, MW, D_, D_, wq_b, 0);
    gemm(WE, wk_w, BK2, MW, D_, D_, wk_b, 0);

    // 3) attention scores, entmax-1.5, head mean
    k_scores<<<S_ * H_, 256, 0, stream>>>(BQ, BK2, SC);
    k_entmax<<<(S_ * H_ * 64 + 255) / 256, 256, 0, stream>>>(SC, S_ * H_ * 64);
    k_headred<<<(int)((latN + 255) / 256), 256, 0, stream>>>(SC, LAT);

    // 4) gated GCN layer 1
    gemm(WE, gc1_w, BH, MW, D_, D_, nullptr, 0);
    {
        dim3 g(D_ / 64, S_);
        k_bgemm_adj<<<g, 256, 0, stream>>>(dep_adj, BH, BO);
        k_bgemm_adj<<<g, 256, 0, stream>>>(LAT, BH, BL);
    }
    gemm(BO, gc1_l1w, BQ, MW, D_, D_, gc1_l1b, 0);
    gemm(BL, gc1_l2w, BQ, MW, D_, D_, gc1_l2b, 1);
    k_gate<<<(int)((NE_ + 255) / 256), 256, 0, stream>>>(BQ, BO, BL, gc1_b, nullptr, BK2, NE_);

    // 5) gated GCN layer 2 + residual
    gemm(BK2, gc2_w, BH, MW, D_, D_, nullptr, 0);
    {
        dim3 g(D_ / 64, S_);
        k_bgemm_adj<<<g, 256, 0, stream>>>(dep_adj, BH, BO);
        k_bgemm_adj<<<g, 256, 0, stream>>>(LAT, BH, BL);
    }
    gemm(BO, gc2_l1w, BQ, MW, D_, D_, gc2_l1b, 0);
    gemm(BL, gc2_l2w, BQ, MW, D_, D_, gc2_l2b, 1);
    k_gate<<<(int)((NE_ + 255) / 256), 256, 0, stream>>>(BQ, BO, BL, gc2_b, WE, BH, NE_);

    // 6) node features
    k_feats<<<(int)((feN + 255) / 256), 256, 0, stream>>>(doc_cls, sent_cls, BH, t_sid, t_index, FE);

    // 7) RelGCN layers
    const float* relW[2]  = { rel1_w, rel2_w };
    const float* relLp[2] = { rel1_loop, rel2_loop };
    const float* relB[2]  = { rel1_b, rel2_b };
    const float* layerIn  = FE;
    float*       layerOut[2] = { H1, H2 };
    for (int l = 0; l < 2; ++l) {
        k_gather_rows<<<R_ * E_, 256, 0, stream>>>(layerIn, edges_src, BO);
        for (int r = 0; r < R_; ++r)
            gemm(BO + (size_t)r * E_ * D_, relW[l] + (size_t)r * D_ * D_,
                 BL + (size_t)r * E_ * D_, E_, D_, D_, nullptr, 0);
        k_zero<<<(R_ * NN_ + 255) / 256, 256, 0, stream>>>(DEG, (size_t)R_ * NN_);
        k_deg<<<(R_ * E_ + 255) / 256, 256, 0, stream>>>(edges_dst, DEG);
        k_zero<<<(int)((feN + 255) / 256), 256, 0, stream>>>(HS, feN);
        k_scatter<<<R_ * E_, 256, 0, stream>>>(BL, edges_dst, DEG, HS);
        gemm(layerIn, relLp[l], LP, NN_, D_, D_, nullptr, 0);
        k_relu_comb<<<(int)((feN + 255) / 256), 256, 0, stream>>>(HS, LP, relB[l], layerOut[l], feN);
        layerIn = layerOut[l];
    }

    // 8) predictor
    k_pred<<<(NN_ * 5 + 255) / 256, 256, 0, stream>>>(FE, H1, H2, pred_w, pred_b, out);
}
